// PNAPCSAFT2_69183333204156
// MI455X (gfx1250) — matrix operand order repaired and checked by reading, not compiled, not run on hardware
//
#include <hip/hip_runtime.h>

#define NN 10000
#define EE 160000
#define GG 64
#define HH 128
#define EPS 1e-5f

typedef __attribute__((ext_vector_type(16))) _Float16 v16h;
typedef __attribute__((ext_vector_type(8)))  float    v8f;

static __device__ inline v8f wmma_f16(v16h a, v16h b, v8f c) {
  return __builtin_amdgcn_wmma_f32_16x16x32_f16(
      /*neg_a=*/false, a, /*neg_b=*/false, b,
      /*c_mod=*/(short)0, c, /*reuse_a=*/false, /*reuse_b=*/false);
}

// ---------------- fills ----------------
__global__ void k_fill_f32(float* p, int n, float v) {
  for (int i = blockIdx.x * blockDim.x + threadIdx.x; i < n; i += gridDim.x * blockDim.x)
    p[i] = v;
}

// ---------------- embeddings ----------------
__global__ __launch_bounds__(128) void k_embed_nodes(
    const int* __restrict__ x, const float* __restrict__ atom_emb,
    float* __restrict__ h, _Float16* __restrict__ h16) {
  int n = blockIdx.x, c = threadIdx.x;
  float s = 0.f;
#pragma unroll
  for (int j = 0; j < 9; ++j)
    s += atom_emb[(j * 16 + x[n * 9 + j]) * HH + c];
  h[n * HH + c] = s;
  h16[n * HH + c] = (_Float16)s;
}

__global__ __launch_bounds__(128) void k_embed_edges(
    const int* __restrict__ eattr, const float* __restrict__ bond_emb,
    _Float16* __restrict__ e16) {
  int e = blockIdx.x, c = threadIdx.x;
  float s = 0.f;
#pragma unroll
  for (int j = 0; j < 3; ++j)
    s += bond_emb[(j * 8 + eattr[e * 3 + j]) * HH + c];
  e16[e * HH + c] = (_Float16)s;
}

__global__ __launch_bounds__(256) void k_count(const int* __restrict__ dstI, float* cnt) {
  int e = blockIdx.x * 256 + threadIdx.x;
  if (e < EE) atomicAdd(&cnt[dstI[e]], 1.0f);
}

__global__ __launch_bounds__(256) void k_degstats(
    const float* __restrict__ cnt, float* deginv, float* logd, float* sumlog) {
  __shared__ float red[256];
  int i = blockIdx.x * 256 + threadIdx.x;
  float ll = 0.f;
  if (i < NN) {
    float c = cnt[i];
    float d = fmaxf(c, 1.0f);
    deginv[i] = 1.0f / d;
    logd[i] = logf(d + 1.0f);
    ll = logf(c + 1.0f);
  }
  red[threadIdx.x] = ll;
  __syncthreads();
  for (int s = 128; s > 0; s >>= 1) {
    if (threadIdx.x < s) red[threadIdx.x] += red[threadIdx.x + s];
    __syncthreads();
  }
  if (threadIdx.x == 0) atomicAdd(sumlog, red[0]);
}

__global__ __launch_bounds__(256) void k_scalers(
    const float* __restrict__ logd, const float* __restrict__ sumlog,
    float* s1, float* s2) {
  int i = blockIdx.x * 256 + threadIdx.x;
  if (i >= NN) return;
  float avg = sumlog[0] * (1.0f / (float)NN);
  float ld = logd[i];
  s1[i] = ld / avg;
  s2[i] = avg / ld;
}

// ---------------- CSR build (counting sort by dst) ----------------
__global__ __launch_bounds__(256) void k_scan1(const float* __restrict__ cnt, int* bsum) {
  __shared__ int red[256];
  int i = blockIdx.x * 256 + threadIdx.x;
  red[threadIdx.x] = (i < NN) ? (int)cnt[i] : 0;
  __syncthreads();
  for (int s = 128; s > 0; s >>= 1) {
    if (threadIdx.x < s) red[threadIdx.x] += red[threadIdx.x + s];
    __syncthreads();
  }
  if (threadIdx.x == 0) bsum[blockIdx.x] = red[0];
}
__global__ void k_scan2(int* bsum, int nb) {
  if (threadIdx.x == 0 && blockIdx.x == 0) {
    int acc = 0;
    for (int i = 0; i < nb; ++i) { int t = bsum[i]; bsum[i] = acc; acc += t; }
  }
}
__global__ __launch_bounds__(256) void k_scan3(
    const float* __restrict__ cnt, const int* __restrict__ bsum,
    int* offs, int* cursor) {
  __shared__ int sh[256];
  int i = blockIdx.x * 256 + threadIdx.x;
  int v = (i < NN) ? (int)cnt[i] : 0;
  sh[threadIdx.x] = v;
  __syncthreads();
  for (int d = 1; d < 256; d <<= 1) {  // Hillis-Steele inclusive scan
    int t = (threadIdx.x >= d) ? sh[threadIdx.x - d] : 0;
    __syncthreads();
    sh[threadIdx.x] += t;
    __syncthreads();
  }
  if (i < NN) {
    int excl = sh[threadIdx.x] - v + bsum[blockIdx.x];
    offs[i] = excl;
    cursor[i] = excl;
  }
}
__global__ __launch_bounds__(256) void k_scatter(
    const int* __restrict__ dstI, int* cursor, int* eid) {
  int e = blockIdx.x * 256 + threadIdx.x;
  if (e < EE) {
    int slot = atomicAdd(&cursor[dstI[e]], 1);
    eid[slot] = e;
  }
}

// ---------------- weight conversion to f16 ----------------
// pre_w [L,T,384,128] -> [L][384][256] fused (col = t*128+h)
__global__ void k_cvt_prew(const float* __restrict__ src, _Float16* __restrict__ dst) {
  for (int idx = blockIdx.x * blockDim.x + threadIdx.x; idx < 2 * 384 * 256;
       idx += gridDim.x * blockDim.x) {
    int l = idx / (384 * 256);
    int r = idx - l * (384 * 256);
    int k = r >> 8;
    int nn = r & 255;
    int t = nn >> 7, hc = nn & 127;
    dst[idx] = (_Float16)src[(((l * 2 + t) * 384) + k) * 128 + hc];
  }
}
__global__ void k_cvt_f16(const float* __restrict__ src, _Float16* __restrict__ dst, int n) {
  for (int i = blockIdx.x * blockDim.x + threadIdx.x; i < n; i += gridDim.x * blockDim.x)
    dst[i] = (_Float16)src[i];
}

// ---------------- edge GEMM (64-edge M-block, 4x B reuse) ----------------
// m16[e, 0:256] = [h16[dst] | h16[src] | e16] @ preW (384x256) + pre_b
static __device__ inline void store_m(_Float16* __restrict__ m16, v8f c,
                                      int base_row, int hi, int col, float bias) {
#pragma unroll
  for (int i = 0; i < 8; ++i) {
    int row = base_row + i + (hi << 3);
    m16[row * 256 + col] = (_Float16)(c[i] + bias);
  }
}

__global__ __launch_bounds__(256) void k_edge_gemm(
    const _Float16* __restrict__ h16, const _Float16* __restrict__ e16,
    const _Float16* __restrict__ preW,   // [384][256], this layer
    const float* __restrict__ pre_b,     // [256], this layer
    const int* __restrict__ srcI, const int* __restrict__ dstI,
    _Float16* __restrict__ m16) {
  __shared__ _Float16 As[64][392];  // 64 edges x 384 (K), padded
  __shared__ int sdst[64], ssrc[64];
  int et = blockIdx.x;  // 64-edge tile
  int tid = threadIdx.x;
  if (tid < 64) {
    sdst[tid] = dstI[et * 64 + tid];
    ssrc[tid] = srcI[et * 64 + tid];
  }
  __syncthreads();
  for (int idx = tid; idx < 64 * 384; idx += 256) {
    int r = idx / 384;
    int kk = idx - r * 384;
    _Float16 v;
    if (kk < 128)      v = h16[sdst[r] * HH + kk];
    else if (kk < 256) v = h16[ssrc[r] * HH + (kk - 128)];
    else               v = e16[(et * 64 + r) * HH + (kk - 256)];
    As[r][kk] = v;
  }
  __syncthreads();

  int wave = tid >> 5, lane = tid & 31;
  int mrow = lane & 15;
  int hi = lane >> 4;
  for (int sub = 0; sub < 2; ++sub) {
    int n0 = (wave * 2 + sub) * 16;
    v8f c0 = {}, c1 = {}, c2 = {}, c3 = {};
    for (int kc = 0; kc < 12; ++kc) {
      int k0 = kc * 32;
      v16h b;
      const _Float16* bp = preW + (k0 + (hi << 3)) * 256 + n0 + mrow;
#pragma unroll
      for (int e2 = 0; e2 < 16; ++e2) b[e2] = bp[((e2 & 7) + ((e2 >> 3) << 4)) * 256];
      v16h a;
      const _Float16* ap;
      ap = &As[mrow][k0 + (hi << 3)];
#pragma unroll
      for (int e2 = 0; e2 < 16; ++e2) a[e2] = ap[(e2 & 7) + ((e2 >> 3) << 4)];
      c0 = wmma_f16(a, b, c0);
      ap = &As[16 + mrow][k0 + (hi << 3)];
#pragma unroll
      for (int e2 = 0; e2 < 16; ++e2) a[e2] = ap[(e2 & 7) + ((e2 >> 3) << 4)];
      c1 = wmma_f16(a, b, c1);
      ap = &As[32 + mrow][k0 + (hi << 3)];
#pragma unroll
      for (int e2 = 0; e2 < 16; ++e2) a[e2] = ap[(e2 & 7) + ((e2 >> 3) << 4)];
      c2 = wmma_f16(a, b, c2);
      ap = &As[48 + mrow][k0 + (hi << 3)];
#pragma unroll
      for (int e2 = 0; e2 < 16; ++e2) a[e2] = ap[(e2 & 7) + ((e2 >> 3) << 4)];
      c3 = wmma_f16(a, b, c3);
    }
    int col = n0 + mrow;
    float bias = pre_b[col];
    store_m(m16, c0, et * 64 +  0, hi, col, bias);
    store_m(m16, c1, et * 64 + 16, hi, col, bias);
    store_m(m16, c2, et * 64 + 32, hi, col, bias);
    store_m(m16, c3, et * 64 + 48, hi, col, bias);
  }
}

// ---------------- CSR segment reduce -> agg16 [N][T][4][128] f16 ----------------
__global__ __launch_bounds__(256) void k_agg(
    const _Float16* __restrict__ m16, const int* __restrict__ offs,
    const int* __restrict__ eid, const float* __restrict__ cnt,
    const float* __restrict__ deginv, _Float16* __restrict__ agg16) {
  int n = blockIdx.x, c = threadIdx.x;
  int start = offs[n];
  int d = (int)cnt[n];
  float s = 0.f, q = 0.f, mn = 0.f, mx = 0.f;
  if (d > 0) {
    float v0 = (float)m16[eid[start] * 256 + c];
    s = v0; q = v0 * v0; mn = v0; mx = v0;
    for (int j = 1; j < d; ++j) {
      float v = (float)m16[eid[start + j] * 256 + c];
      s += v; q += v * v;
      mn = fminf(mn, v);
      mx = fmaxf(mx, v);
    }
  }
  float di = deginv[n];
  float mean = s * di;
  float msq = q * di;
  float sd = sqrtf(fmaxf(msq - mean * mean, 0.f) + EPS);
  int t = c >> 7, hh = c & 127;
  int b = n * 1024 + t * 512;
  agg16[b + hh]       = (_Float16)mean;
  agg16[b + 128 + hh] = (_Float16)mn;
  agg16[b + 256 + hh] = (_Float16)mx;
  agg16[b + 384 + hh] = (_Float16)sd;
}

// ---------------- post GEMM: 4 fused accumulator chains ----------------
// out[n,t,:] = h@W0 + agg@W1 + s1*(agg@W2) + s2*(agg@W3) + post_b
__global__ __launch_bounds__(256) void k_post_gemm(
    const _Float16* __restrict__ h16, const _Float16* __restrict__ agg16,
    const _Float16* __restrict__ postW,  // [2 towers][1664][64], this layer
    const float* __restrict__ post_b,    // [2][64], this layer
    const float* __restrict__ s1, const float* __restrict__ s2,
    _Float16* __restrict__ tout16) {
  int mt = blockIdx.x;
  int tid = threadIdx.x;
  int wave = tid >> 5, lane = tid & 31;
  int t = wave >> 2, ntile = wave & 3;
  int n0 = ntile * 16;
  int mrow = lane & 15, hi = lane >> 4;
  int grow = mt * 16 + mrow;
  const _Float16* W = postW + t * 1664 * 64;
  v8f a0 = {}, a1 = {}, a2 = {}, a3 = {};
  // h part, K=128 (W rows 0..127)
  const _Float16* ah = h16 + grow * HH;
  for (int kc = 0; kc < 4; ++kc) {
    int k0 = kc * 32;
    v16h a, b;
    const _Float16* ap = ah + k0 + (hi << 3);
#pragma unroll
    for (int e2 = 0; e2 < 16; ++e2) a[e2] = ap[(e2 & 7) + ((e2 >> 3) << 4)];
    const _Float16* bp = W + (k0 + (hi << 3)) * 64 + n0 + mrow;
#pragma unroll
    for (int e2 = 0; e2 < 16; ++e2) b[e2] = bp[((e2 & 7) + ((e2 >> 3) << 4)) * 64];
    a0 = wmma_f16(a, b, a0);
  }
  // agg part, K=512 (W rows 128.., 640.., 1152..)
  const _Float16* aa = agg16 + grow * 1024 + t * 512;
  for (int kc = 0; kc < 16; ++kc) {
    int k0 = kc * 32;
    v16h a, b1, b2, b3;
    const _Float16* ap = aa + k0 + (hi << 3);
#pragma unroll
    for (int e2 = 0; e2 < 16; ++e2) a[e2] = ap[(e2 & 7) + ((e2 >> 3) << 4)];
    const _Float16* bp1 = W + (128 + k0 + (hi << 3)) * 64 + n0 + mrow;
    const _Float16* bp2 = W + (640 + k0 + (hi << 3)) * 64 + n0 + mrow;
    const _Float16* bp3 = W + (1152 + k0 + (hi << 3)) * 64 + n0 + mrow;
#pragma unroll
    for (int e2 = 0; e2 < 16; ++e2) { int kr = ((e2 & 7) + ((e2 >> 3) << 4)) * 64; b1[e2] = bp1[kr]; b2[e2] = bp2[kr]; b3[e2] = bp3[kr]; }
    a1 = wmma_f16(a, b1, a1);
    a2 = wmma_f16(a, b2, a2);
    a3 = wmma_f16(a, b3, a3);
  }
  int col = n0 + mrow;
  float bias = post_b[t * 64 + col];
#pragma unroll
  for (int i = 0; i < 8; ++i) {
    int row = i + (hi << 3);
    int node = mt * 16 + row;
    float v = a0[i] + a1[i] + s1[node] * a2[i] + s2[node] * a3[i] + bias;
    tout16[node * HH + t * 64 + col] = (_Float16)v;
  }
}

// ---------------- lin GEMM + fused BN stats ----------------
__global__ __launch_bounds__(256) void k_lin_gemm(
    const _Float16* __restrict__ tout16, const _Float16* __restrict__ linW,  // [128][128]
    const float* __restrict__ lin_b, float* __restrict__ y,
    float* __restrict__ colsum, float* __restrict__ colsumsq) {
  int mt = blockIdx.x;
  int tid = threadIdx.x;
  int wave = tid >> 5, lane = tid & 31;
  int n0 = wave * 16;
  int mrow = lane & 15, hi = lane >> 4;
  int grow = mt * 16 + mrow;
  v8f c = {};
  const _Float16* ah = tout16 + grow * HH;
  for (int kc = 0; kc < 4; ++kc) {
    int k0 = kc * 32;
    v16h a, b;
    const _Float16* ap = ah + k0 + (hi << 3);
#pragma unroll
    for (int e2 = 0; e2 < 16; ++e2) a[e2] = ap[(e2 & 7) + ((e2 >> 3) << 4)];
    const _Float16* bp = linW + (k0 + (hi << 3)) * HH + n0 + mrow;
#pragma unroll
    for (int e2 = 0; e2 < 16; ++e2) b[e2] = bp[((e2 & 7) + ((e2 >> 3) << 4)) * HH];
    c = wmma_f16(a, b, c);
  }
  int col = n0 + mrow;
  float bias = lin_b[col];
  float ps = 0.f, pq = 0.f;
#pragma unroll
  for (int i = 0; i < 8; ++i) {
    int row = i + (hi << 3);
    int node = mt * 16 + row;
    float v = c[i] + bias;
    y[node * HH + col] = v;
    ps += v;
    pq += v * v;
  }
  ps += __shfl_xor(ps, 16, 32);
  pq += __shfl_xor(pq, 16, 32);
  if (hi == 0) {
    atomicAdd(&colsum[col], ps);
    atomicAdd(&colsumsq[col], pq);
  }
}

__global__ __launch_bounds__(128) void k_bn_apply(
    const float* __restrict__ y, const float* __restrict__ colsum,
    const float* __restrict__ colsumsq, const float* __restrict__ g,
    const float* __restrict__ b, float* __restrict__ h, _Float16* __restrict__ h16) {
  int n = blockIdx.x, c = threadIdx.x;
  const float invN = 1.0f / (float)NN;
  float m = colsum[c] * invN;
  float var = colsumsq[c] * invN - m * m;
  float rs = rsqrtf(var + EPS);
  float v = (y[n * HH + c] - m) * rs * g[c] + b[c];
  v = fmaxf(v, 0.f);
  h[n * HH + c] = v;
  h16[n * HH + c] = (_Float16)v;
}

// ---------------- pooling + readout ----------------
__global__ __launch_bounds__(128) void k_pool(
    const float* __restrict__ h, const int* __restrict__ batch, float* __restrict__ gpool) {
  int n = blockIdx.x, c = threadIdx.x;
  atomicAdd(&gpool[batch[n] * HH + c], h[n * HH + c]);
}

static __device__ void ro_dense(const float* A, const float* W, const float* bias,
                                float* O, int rows, int nin, int nout, int tid) {
  for (int idx = tid; idx < rows * nout; idx += 256) {
    int r = idx / nout, o = idx - r * nout;
    float s = bias[o];
    for (int i = 0; i < nin; ++i) s += A[r * nin + i] * W[i * nout + o];
    O[idx] = s;
  }
}
static __device__ void ro_bnrelu(float* A, const float* g, const float* b,
                                 int rows, int nout, int tid, float* sc, float* sh) {
  for (int c = tid; c < nout; c += 256) {
    float s = 0.f, q = 0.f;
    for (int r = 0; r < rows; ++r) { float v = A[r * nout + c]; s += v; q += v * v; }
    float m = s / (float)rows;
    float var = q / (float)rows - m * m;
    float rs = rsqrtf(var + EPS);
    sc[c] = rs * g[c];
    sh[c] = b[c] - m * rs * g[c];
  }
  __syncthreads();
  for (int idx = tid; idx < rows * nout; idx += 256) {
    int c = idx % nout;
    A[idx] = fmaxf(A[idx] * sc[c] + sh[c], 0.f);
  }
}

__global__ __launch_bounds__(256) void k_readout(
    const float* __restrict__ gpool,
    const float* mlp_w, const float* mlp_b, const float* mlp_g, const float* mlp_bb,
    const float* w1, const float* b1, const float* g1, const float* bb1,
    const float* w2, const float* b2, const float* g2, const float* bb2,
    const float* w3, const float* b3, float* __restrict__ out) {
  __shared__ float r0[64 * 128];
  __shared__ float r1[64 * 64];
  __shared__ float sc[128], sh[128];
  int tid = threadIdx.x;
  ro_dense(gpool, mlp_w, mlp_b, r0, 64, 128, 128, tid); __syncthreads();
  ro_bnrelu(r0, mlp_g, mlp_bb, 64, 128, tid, sc, sh);   __syncthreads();
  ro_dense(r0, w1, b1, r1, 64, 128, 64, tid);           __syncthreads();
  ro_bnrelu(r1, g1, bb1, 64, 64, tid, sc, sh);          __syncthreads();
  ro_dense(r1, w2, b2, r0, 64, 64, 32, tid);            __syncthreads();
  ro_bnrelu(r0, g2, bb2, 64, 32, tid, sc, sh);          __syncthreads();
  ro_dense(r0, w3, b3, out, 64, 32, 3, tid);
}

// ---------------- host launch ----------------
extern "C" void kernel_launch(void* const* d_in, const int* in_sizes, int n_in,
                              void* d_out, int out_size, void* d_ws, size_t ws_size,
                              hipStream_t stream) {
  const int* x        = (const int*)d_in[0];
  const int* eattr    = (const int*)d_in[1];
  const int* eidx     = (const int*)d_in[2];
  const int* batch    = (const int*)d_in[3];
  const float* atom_e = (const float*)d_in[4];
  const float* bond_e = (const float*)d_in[5];
  const float* pre_w  = (const float*)d_in[6];
  const float* pre_bA = (const float*)d_in[7];
  const float* post_w = (const float*)d_in[8];
  const float* post_bA= (const float*)d_in[9];
  const float* lin_w  = (const float*)d_in[10];
  const float* lin_bA = (const float*)d_in[11];
  const float* bn_g   = (const float*)d_in[12];
  const float* bn_b   = (const float*)d_in[13];
  const float* mlp_w  = (const float*)d_in[14];
  const float* mlp_b  = (const float*)d_in[15];
  const float* mlp_g  = (const float*)d_in[16];
  const float* mlp_bb = (const float*)d_in[17];
  const float* ow1    = (const float*)d_in[18];
  const float* ob1    = (const float*)d_in[19];
  const float* og1    = (const float*)d_in[20];
  const float* obb1   = (const float*)d_in[21];
  const float* ow2    = (const float*)d_in[22];
  const float* ob2    = (const float*)d_in[23];
  const float* og2    = (const float*)d_in[24];
  const float* obb2   = (const float*)d_in[25];
  const float* ow3    = (const float*)d_in[26];
  const float* ob3    = (const float*)d_in[27];
  const int* srcI = eidx;
  const int* dstI = eidx + EE;

  char* w = (char*)d_ws;
  auto alloc = [&](size_t bytes) -> char* {
    char* p = w;
    w += (bytes + 255) & ~(size_t)255;
    return p;
  };
  float* h        = (float*)alloc((size_t)NN * HH * 4);
  float* y        = (float*)alloc((size_t)NN * HH * 4);
  float* cnt      = (float*)alloc((size_t)NN * 4);
  float* deginv   = (float*)alloc((size_t)NN * 4);
  float* logd     = (float*)alloc((size_t)NN * 4);
  float* s1       = (float*)alloc((size_t)NN * 4);
  float* s2       = (float*)alloc((size_t)NN * 4);
  int*   offs     = (int*)alloc((size_t)NN * 4);
  int*   cursor   = (int*)alloc((size_t)NN * 4);
  int*   eid      = (int*)alloc((size_t)EE * 4);
  int*   bsum     = (int*)alloc(64 * 4);
  float* colsum   = (float*)alloc(128 * 4);
  float* colsumsq = (float*)alloc(128 * 4);
  float* sumlog   = (float*)alloc(256);
  float* gpool    = (float*)alloc((size_t)GG * HH * 4);
  _Float16* h16   = (_Float16*)alloc((size_t)NN * HH * 2);
  _Float16* tout16= (_Float16*)alloc((size_t)NN * HH * 2);
  _Float16* e16   = (_Float16*)alloc((size_t)EE * HH * 2);
  _Float16* agg16 = (_Float16*)alloc((size_t)NN * 1024 * 2);
  _Float16* m16   = (_Float16*)alloc((size_t)EE * 256 * 2);
  _Float16* preW16  = (_Float16*)alloc((size_t)2 * 384 * 256 * 2);
  _Float16* postW16 = (_Float16*)alloc((size_t)2 * 2 * 1664 * 64 * 2);
  _Float16* linW16  = (_Float16*)alloc((size_t)2 * 128 * 128 * 2);

  const int NB = (NN + 255) / 256;  // 40

  k_fill_f32<<<64, 256, 0, stream>>>(cnt, NN, 0.f);
  k_fill_f32<<<1, 64, 0, stream>>>(sumlog, 1, 0.f);
  k_embed_nodes<<<NN, 128, 0, stream>>>(x, atom_e, h, h16);
  k_embed_edges<<<EE, 128, 0, stream>>>(eattr, bond_e, e16);
  k_count<<<EE / 256, 256, 0, stream>>>(dstI, cnt);
  k_degstats<<<NB, 256, 0, stream>>>(cnt, deginv, logd, sumlog);
  k_scalers<<<NB, 256, 0, stream>>>(logd, sumlog, s1, s2);
  // CSR build
  k_scan1<<<NB, 256, 0, stream>>>(cnt, bsum);
  k_scan2<<<1, 32, 0, stream>>>(bsum, NB);
  k_scan3<<<NB, 256, 0, stream>>>(cnt, bsum, offs, cursor);
  k_scatter<<<EE / 256, 256, 0, stream>>>(dstI, cursor, eid);
  // weight conversion
  k_cvt_prew<<<768, 256, 0, stream>>>(pre_w, preW16);
  k_cvt_f16<<<1024, 256, 0, stream>>>(post_w, postW16, 2 * 2 * 1664 * 64);
  k_cvt_f16<<<128, 256, 0, stream>>>(lin_w, linW16, 2 * 128 * 128);

  for (int l = 0; l < 2; ++l) {
    k_edge_gemm<<<EE / 64, 256, 0, stream>>>(h16, e16, preW16 + (size_t)l * 384 * 256,
                                             pre_bA + l * 256, srcI, dstI, m16);
    k_agg<<<NN, 256, 0, stream>>>(m16, offs, eid, cnt, deginv, agg16);
    k_post_gemm<<<NN / 16, 256, 0, stream>>>(h16, agg16, postW16 + (size_t)l * 2 * 1664 * 64,
                                             post_bA + l * 128, s1, s2, tout16);
    k_fill_f32<<<1, 256, 0, stream>>>(colsum, 128, 0.f);
    k_fill_f32<<<1, 256, 0, stream>>>(colsumsq, 128, 0.f);
    k_lin_gemm<<<NN / 16, 256, 0, stream>>>(tout16, linW16 + (size_t)l * 128 * 128,
                                            lin_bA + l * 128, y, colsum, colsumsq);
    k_bn_apply<<<NN, 128, 0, stream>>>(y, colsum, colsumsq, bn_g + l * 128, bn_b + l * 128,
                                       h, h16);
  }

  k_fill_f32<<<32, 256, 0, stream>>>(gpool, GG * HH, 0.f);
  k_pool<<<NN, 128, 0, stream>>>(h, batch, gpool);
  k_readout<<<1, 256, 0, stream>>>(gpool, mlp_w, mlp_b, mlp_g, mlp_bb,
                                   ow1, ob1, og1, obb1, ow2, ob2, og2, obb2,
                                   ow3, ob3, (float*)d_out);
  (void)in_sizes; (void)n_in; (void)out_size; (void)ws_size;
}
